// BitLinear_41523743818396
// MI455X (gfx1250) — matrix operand order repaired and checked by reading, not compiled, not run on hardware
//
#include <hip/hip_runtime.h>

// ---------------------------------------------------------------------------
// BitLinear: out = input @ sign(W)^T + bias     (M=8192, N=4096, K=4096)
//
// Path A (default, needs ~168MB of d_ws):
//   1) prep_weight : W f32 -> sign bf16 (+/-1), row-major natural K order
//   2) prep_input  : A f32 -> bf16 hi + bf16 lo, K-permuted per 32-block so
//                    each lane's WMMA A-fragment is 32 contiguous bytes
//   3) gemm        : double-buffered async global->LDS staging
//                    (global_load_async_to_lds_b128 / ASYNCcnt), pure
//                    v_wmma_f32_16x16x32_bf16 inner loop (2 WMMA per tile:
//                    hi and lo into the same f32 accumulator ~= fp32 accuracy)
// Path B (fallback if ws too small): fused convert+GEMM kernel.
// ---------------------------------------------------------------------------

typedef __attribute__((ext_vector_type(16))) __bf16 v16bf;
typedef __attribute__((ext_vector_type(8)))  float  v8f;
typedef __attribute__((ext_vector_type(4)))  int    v4i;

typedef __attribute__((address_space(1))) v4i* gv4i_ptr;   // global
typedef __attribute__((address_space(3))) v4i* lv4i_ptr;   // LDS

#define K_DIM 4096
#define N_DIM 4096
#define BM 128
#define BN 128
#define BK 32
#define LDS_STRIDE 40   // ushorts per LDS tile row: 32 payload + 8 pad (80B)

#if defined(__has_builtin)
#if __has_builtin(__builtin_amdgcn_global_load_async_to_lds_b128)
#define HAVE_ASYNC_LDS 1
#endif
#endif
#ifndef HAVE_ASYNC_LDS
#define HAVE_ASYNC_LDS 0
#endif

__device__ __forceinline__ unsigned short f2bf_bits(float x) {
  union { float f; unsigned u; } v; v.f = x;
  unsigned r = v.u + 0x7FFFu + ((v.u >> 16) & 1u);   // round-to-nearest-even
  return (unsigned short)(r >> 16);
}
__device__ __forceinline__ float bf_bits2f(unsigned short h) {
  union { float f; unsigned u; } v; v.u = ((unsigned)h) << 16;
  return v.f;
}
__device__ __forceinline__ unsigned short sign_bf_bits(float x) {
  return x > 0.0f ? 0x3F80u : (x < 0.0f ? 0xBF80u : 0x0000u);
}

union FragU { v16bf v; uint4 q[2]; };

// 16-byte global->LDS copy: async (ASYNCcnt) when available.
__device__ __forceinline__ void copy16(const unsigned short* g, unsigned short* l) {
#if HAVE_ASYNC_LDS
  __builtin_amdgcn_global_load_async_to_lds_b128(
      (gv4i_ptr)(unsigned short*)g, (lv4i_ptr)l, 0, 0);
#else
  *(uint4*)l = *(const uint4*)g;
#endif
}
__device__ __forceinline__ void wait_copies() {
#if HAVE_ASYNC_LDS
#if defined(__has_builtin) && __has_builtin(__builtin_amdgcn_s_wait_asynccnt)
  __builtin_amdgcn_s_wait_asynccnt(0);
#else
  asm volatile("s_wait_asynccnt 0x0" ::: "memory");
#endif
#endif
}

// ---------------------------------------------------------------------------
// Prep kernels (Path A)
// ---------------------------------------------------------------------------
__global__ __launch_bounds__(256) void prep_weight_kernel(
    const float* __restrict__ w, unsigned short* __restrict__ wsgn) {
  const size_t i = ((size_t)blockIdx.x * 256 + threadIdx.x) * 8;
  const float4 a = *(const float4*)(w + i);
  const float4 b = *(const float4*)(w + i + 4);
  uint4 p;
  p.x = (unsigned)sign_bf_bits(a.x) | ((unsigned)sign_bf_bits(a.y) << 16);
  p.y = (unsigned)sign_bf_bits(a.z) | ((unsigned)sign_bf_bits(a.w) << 16);
  p.z = (unsigned)sign_bf_bits(b.x) | ((unsigned)sign_bf_bits(b.y) << 16);
  p.w = (unsigned)sign_bf_bits(b.z) | ((unsigned)sign_bf_bits(b.w) << 16);
  *(uint4*)(wsgn + i) = p;
}

__global__ __launch_bounds__(256) void prep_input_kernel(
    const float* __restrict__ in, unsigned short* __restrict__ ahi,
    unsigned short* __restrict__ alo) {
  const size_t i = ((size_t)blockIdx.x * 256 + threadIdx.x) * 8;
  const int k = (int)(i & (K_DIM - 1));
  // 8-block permutation [0,2,1,3] inside each 32-block -> fragment-ready order
  const int blk8 = (k >> 3) & 3;
  const int dest = ((blk8 & 1) << 1) | (blk8 >> 1);
  const size_t di = (i & ~(size_t)31) | ((size_t)dest << 3);

  const float4 a = *(const float4*)(in + i);
  const float4 b = *(const float4*)(in + i + 4);
  float xs[8] = {a.x, a.y, a.z, a.w, b.x, b.y, b.z, b.w};
  unsigned short hi[8], lo[8];
#pragma unroll
  for (int e = 0; e < 8; ++e) {
    hi[e] = f2bf_bits(xs[e]);
    lo[e] = f2bf_bits(xs[e] - bf_bits2f(hi[e]));
  }
  uint4 ph, pl;
  ph.x = (unsigned)hi[0] | ((unsigned)hi[1] << 16);
  ph.y = (unsigned)hi[2] | ((unsigned)hi[3] << 16);
  ph.z = (unsigned)hi[4] | ((unsigned)hi[5] << 16);
  ph.w = (unsigned)hi[6] | ((unsigned)hi[7] << 16);
  pl.x = (unsigned)lo[0] | ((unsigned)lo[1] << 16);
  pl.y = (unsigned)lo[2] | ((unsigned)lo[3] << 16);
  pl.z = (unsigned)lo[4] | ((unsigned)lo[5] << 16);
  pl.w = (unsigned)lo[6] | ((unsigned)lo[7] << 16);
  *(uint4*)(ahi + di) = ph;
  *(uint4*)(alo + di) = pl;
}

// ---------------------------------------------------------------------------
// GEMM (Path A): double-buffered async staging + WMMA
// ---------------------------------------------------------------------------
__global__ __launch_bounds__(256) void bitlinear_gemm_kernel(
    const unsigned short* __restrict__ ahi, const unsigned short* __restrict__ alo,
    const unsigned short* __restrict__ wsgn, const float* __restrict__ bias,
    float* __restrict__ out) {
  __shared__ __align__(16) unsigned short sAhi[2][BM * LDS_STRIDE];
  __shared__ __align__(16) unsigned short sAlo[2][BM * LDS_STRIDE];
  __shared__ __align__(16) unsigned short sB  [2][BN * LDS_STRIDE];

  const int tid  = threadIdx.x;
  const int wave = tid >> 5;
  const int lane = tid & 31;
  const int half = lane >> 4;
  const int lr   = lane & 15;
  const int wm0  = (wave & 3) * 32;
  const int wn0  = (wave >> 2) * 64;
  const int m0   = blockIdx.y * BM;
  const int n0   = blockIdx.x * BN;

  v8f acc[2][4];
#pragma unroll
  for (int mt = 0; mt < 2; ++mt)
#pragma unroll
    for (int nt = 0; nt < 4; ++nt)
      acc[mt][nt] = v8f{};

  // stage one K-slab (3 tiles x 8KB) into LDS buffer `buf`
  auto stage = [&](int buf, int k0) {
#pragma unroll
    for (int i = 0; i < 2; ++i) {
      const int c   = tid + i * 256;      // 0..511 16B chunks per tile
      const int row = c >> 2;
      const int seg = c & 3;
      const size_t ga = (size_t)(m0 + row) * K_DIM + k0 + seg * 8;
      const size_t gb = (size_t)(n0 + row) * K_DIM + k0 + seg * 8;
      const int    lo = row * LDS_STRIDE + seg * 8;
      copy16(ahi  + ga, &sAhi[buf][lo]);
      copy16(alo  + ga, &sAlo[buf][lo]);
      copy16(wsgn + gb, &sB  [buf][lo]);
    }
  };

  constexpr int NK = K_DIM / BK;   // 128
  stage(0, 0);

  for (int kt = 0; kt < NK; ++kt) {
    const int buf = kt & 1;
    wait_copies();
    __syncthreads();
    if (kt + 1 < NK) stage(buf ^ 1, (kt + 1) * BK);

    FragU bf[4];
#pragma unroll
    for (int nt = 0; nt < 4; ++nt) {
      const unsigned short* p = &sB[buf][(wn0 + nt * 16 + lr) * LDS_STRIDE + half * 8];
      bf[nt].q[0] = *(const uint4*)p;
      bf[nt].q[1] = *(const uint4*)(p + 16);
    }
#pragma unroll
    for (int mt = 0; mt < 2; ++mt) {
      FragU ah, al;
      const unsigned short* ph = &sAhi[buf][(wm0 + mt * 16 + lr) * LDS_STRIDE + half * 16];
      const unsigned short* pl = &sAlo[buf][(wm0 + mt * 16 + lr) * LDS_STRIDE + half * 16];
      ah.q[0] = *(const uint4*)ph;  ah.q[1] = *(const uint4*)(ph + 8);
      al.q[0] = *(const uint4*)pl;  al.q[1] = *(const uint4*)(pl + 8);
#pragma unroll
      for (int nt = 0; nt < 4; ++nt) {
        acc[mt][nt] = __builtin_amdgcn_wmma_f32_16x16x32_bf16(
            false, ah.v, false, bf[nt].v, (short)0, acc[mt][nt], false, false);
        acc[mt][nt] = __builtin_amdgcn_wmma_f32_16x16x32_bf16(
            false, al.v, false, bf[nt].v, (short)0, acc[mt][nt], false, false);
      }
    }
  }

#pragma unroll
  for (int mt = 0; mt < 2; ++mt) {
#pragma unroll
    for (int nt = 0; nt < 4; ++nt) {
      const int col = n0 + wn0 + nt * 16 + lr;
      const float bv = bias[col];
      const int rbase = m0 + wm0 + mt * 16 + half * 8;
#pragma unroll
      for (int j = 0; j < 8; ++j)
        out[(size_t)(rbase + j) * N_DIM + col] = acc[mt][nt][j] + bv;
    }
  }
}

// ---------------------------------------------------------------------------
// Path B fallback: fused convert + GEMM
// ---------------------------------------------------------------------------
__global__ __launch_bounds__(256) void bitlinear_fused_kernel(
    const float* __restrict__ input, const float* __restrict__ weight,
    const float* __restrict__ bias, float* __restrict__ out) {
  __shared__ __align__(16) unsigned short sAhi[BM * LDS_STRIDE];
  __shared__ __align__(16) unsigned short sAlo[BM * LDS_STRIDE];
  __shared__ __align__(16) unsigned short sB  [BN * LDS_STRIDE];

  const int tid  = threadIdx.x;
  const int wave = tid >> 5;
  const int lane = tid & 31;
  const int half = lane >> 4;
  const int lr   = lane & 15;
  const int wm0  = (wave & 3) * 32;
  const int wn0  = (wave >> 2) * 64;
  const int m0   = blockIdx.y * BM;
  const int n0   = blockIdx.x * BN;

  v8f acc[2][4];
#pragma unroll
  for (int mt = 0; mt < 2; ++mt)
#pragma unroll
    for (int nt = 0; nt < 4; ++nt)
      acc[mt][nt] = v8f{};

  for (int k0 = 0; k0 < K_DIM; k0 += BK) {
#pragma unroll
    for (int i = 0; i < 4; ++i) {
      const int idx = tid + i * 256;
      const int row = idx >> 3;
      const int kg  = idx & 7;
      const float* ap = input  + (size_t)(m0 + row) * K_DIM + k0 + kg * 4;
      const float* wp = weight + (size_t)(n0 + row) * K_DIM + k0 + kg * 4;
      const float4 a4 = *(const float4*)ap;
      const float4 w4 = *(const float4*)wp;
      float xs[4] = {a4.x, a4.y, a4.z, a4.w};
      unsigned short hi[4], lo[4];
#pragma unroll
      for (int e = 0; e < 4; ++e) {
        hi[e] = f2bf_bits(xs[e]);
        lo[e] = f2bf_bits(xs[e] - bf_bits2f(hi[e]));
      }
      const int blk     = kg >> 1;
      const int destblk = ((blk & 1) << 1) | ((blk >> 1) & 1);
      const int posA    = destblk * 8 + (kg & 1) * 4;
      const int posB    = kg * 4;
      uint2 ph, pl, pw;
      ph.x = (unsigned)hi[0] | ((unsigned)hi[1] << 16);
      ph.y = (unsigned)hi[2] | ((unsigned)hi[3] << 16);
      pl.x = (unsigned)lo[0] | ((unsigned)lo[1] << 16);
      pl.y = (unsigned)lo[2] | ((unsigned)lo[3] << 16);
      pw.x = (unsigned)sign_bf_bits(w4.x) | ((unsigned)sign_bf_bits(w4.y) << 16);
      pw.y = (unsigned)sign_bf_bits(w4.z) | ((unsigned)sign_bf_bits(w4.w) << 16);
      *(uint2*)&sAhi[row * LDS_STRIDE + posA] = ph;
      *(uint2*)&sAlo[row * LDS_STRIDE + posA] = pl;
      *(uint2*)&sB  [row * LDS_STRIDE + posB] = pw;
    }
    __syncthreads();

    FragU bf[4];
#pragma unroll
    for (int nt = 0; nt < 4; ++nt) {
      const unsigned short* p = &sB[(wn0 + nt * 16 + lr) * LDS_STRIDE + half * 8];
      bf[nt].q[0] = *(const uint4*)p;
      bf[nt].q[1] = *(const uint4*)(p + 16);
    }
#pragma unroll
    for (int mt = 0; mt < 2; ++mt) {
      FragU ah, al;
      const unsigned short* ph = &sAhi[(wm0 + mt * 16 + lr) * LDS_STRIDE + half * 16];
      const unsigned short* pl = &sAlo[(wm0 + mt * 16 + lr) * LDS_STRIDE + half * 16];
      ah.q[0] = *(const uint4*)ph;  ah.q[1] = *(const uint4*)(ph + 8);
      al.q[0] = *(const uint4*)pl;  al.q[1] = *(const uint4*)(pl + 8);
#pragma unroll
      for (int nt = 0; nt < 4; ++nt) {
        acc[mt][nt] = __builtin_amdgcn_wmma_f32_16x16x32_bf16(
            false, ah.v, false, bf[nt].v, (short)0, acc[mt][nt], false, false);
        acc[mt][nt] = __builtin_amdgcn_wmma_f32_16x16x32_bf16(
            false, al.v, false, bf[nt].v, (short)0, acc[mt][nt], false, false);
      }
    }
    __syncthreads();
  }

#pragma unroll
  for (int mt = 0; mt < 2; ++mt) {
#pragma unroll
    for (int nt = 0; nt < 4; ++nt) {
      const int col = n0 + wn0 + nt * 16 + lr;
      const float bv = bias[col];
      const int rbase = m0 + wm0 + mt * 16 + half * 8;
#pragma unroll
      for (int j = 0; j < 8; ++j)
        out[(size_t)(rbase + j) * N_DIM + col] = acc[mt][nt][j] + bv;
    }
  }
}

// ---------------------------------------------------------------------------
extern "C" void kernel_launch(void* const* d_in, const int* in_sizes, int n_in,
                              void* d_out, int out_size, void* d_ws, size_t ws_size,
                              hipStream_t stream) {
  (void)n_in; (void)out_size;
  const float* input  = (const float*)d_in[0];
  const float* weight = (const float*)d_in[1];
  const float* bias   = (const float*)d_in[2];
  float* out = (float*)d_out;

  const int M = in_sizes[0] / K_DIM;                 // 8192
  const size_t nW = (size_t)N_DIM * K_DIM;           // 16.7M
  const size_t nA = (size_t)M * K_DIM;               // 33.5M
  const size_t needed = (nW + 2 * nA) * sizeof(unsigned short);  // ~168 MB

  dim3 grid(N_DIM / BN, M / BM);

  if (ws_size >= needed) {
    unsigned short* wsgn = (unsigned short*)d_ws;
    unsigned short* ahi  = wsgn + nW;
    unsigned short* alo  = ahi + nA;
    prep_weight_kernel<<<dim3((unsigned)(nW / 2048)), 256, 0, stream>>>(weight, wsgn);
    prep_input_kernel <<<dim3((unsigned)(nA / 2048)), 256, 0, stream>>>(input, ahi, alo);
    bitlinear_gemm_kernel<<<grid, 256, 0, stream>>>(ahi, alo, wsgn, bias, out);
  } else {
    bitlinear_fused_kernel<<<grid, 256, 0, stream>>>(input, weight, bias, out);
  }
}
